// AtomDecoder_79370995630835
// MI455X (gfx1250) — matrix operand order repaired and checked by reading, not compiled, not run on hardware
//
#include <hip/hip_runtime.h>
#include <hip/hip_bf16.h>
#include <math.h>

typedef __attribute__((ext_vector_type(16))) _Float16 v16h;
typedef __attribute__((ext_vector_type(8)))  _Float16 v8h;
typedef __attribute__((ext_vector_type(8)))  float    v8f;

#define BB 16
#define AA 512
#define MM (BB*AA)        // 8192 rows
#define DA 256            // d_atom
#define DI 512            // d_inner
#define NH 8              // heads
#define PP 64             // headdim
#define RR 2              // mimo rank
#define NSTATE 32         // d_state
#define DPROJ 1160        // z(512) | xc(512) | B(64) | C(64) | dt(8)
#define NPAD_IN 1184      // DPROJ padded to 32
#define DCONV 4

__device__ __forceinline__ float siluf(float x){ return x / (1.f + expf(-x)); }

// ---------------------------------------------------------------------------
// Convert fp32 weight [K,N] -> f16 transposed [Npad,K] (zero pad rows >= N)
// ---------------------------------------------------------------------------
__global__ void k_w2h_t(const float* __restrict__ W, _Float16* __restrict__ WT,
                        int K, int N, int Npad){
  int idx = blockIdx.x * blockDim.x + threadIdx.x;
  if (idx >= Npad * K) return;
  int n = idx / K, k = idx - n * K;
  float v = (n < N) ? W[(size_t)k * N + n] : 0.f;
  WT[idx] = (_Float16)v;
}

// ---------------------------------------------------------------------------
// Pre-RMSNorm over D_ATOM -> f16 activations
// ---------------------------------------------------------------------------
__global__ void k_rms16(const float* __restrict__ x, const float* __restrict__ w,
                        _Float16* __restrict__ out){
  __shared__ float red[256];
  int row = blockIdx.x, t = threadIdx.x;
  float v = x[(size_t)row * DA + t];
  red[t] = v * v; __syncthreads();
  for (int s = 128; s > 0; s >>= 1){ if (t < s) red[t] += red[t + s]; __syncthreads(); }
  float scale = rsqrtf(red[0] / (float)DA + 1e-6f);
  out[(size_t)row * DA + t] = (_Float16)(v * scale * w[t]);
}

// ---------------------------------------------------------------------------
// WMMA GEMM, 2x2 register-blocked: one wave32 computes a 32x32 output tile
// (4 accumulators; each A fragment reused for 2 N-tiles and vice versa).
// out[M,N] = A[M,K](f16) @ BT[Npad,K](f16)^T (+Cin)(+bias)(optional GELU)
// C/D fragment layout per CDNA5 ISA 7.12.2; A/B use contiguous k = 16*half + s.
// ---------------------------------------------------------------------------
__global__ void k_wmma_gemm(const _Float16* __restrict__ Af, const _Float16* __restrict__ Bt,
                            int M, int N, int K,
                            const float* __restrict__ Cin, const float* __restrict__ bias,
                            float* __restrict__ outF, _Float16* __restrict__ outH,
                            int act_gelu){
  int lane = threadIdx.x;
  int hl = lane >> 4;           // half-wave select
  int l16 = lane & 15;
  int mbase = blockIdx.y * 32, nbase = blockIdx.x * 32;
  const _Float16* Arow[2] = {
    Af + (size_t)(mbase + l16) * K,
    Af + (size_t)(mbase + 16 + l16) * K };
  const _Float16* Brow[2] = {
    Bt + (size_t)(nbase + l16) * K + hl * 16,
    Bt + (size_t)(nbase + 16 + l16) * K + hl * 16 };
  v8f acc[2][2] = {};
  for (int k0 = 0; k0 < K; k0 += 32){
    v16h a[2], b[2];
#pragma unroll
    for (int mi = 0; mi < 2; mi++){
      // A 16x32: lanes 0-15 elems0-15=K0..15; lanes16-31 K16..31
      v8h alo = *(const v8h*)(Arow[mi] + k0 + hl * 16);
      v8h ahi = *(const v8h*)(Arow[mi] + k0 + 8 + hl * 16);
#pragma unroll
      for (int i = 0; i < 8; i++){ a[mi][i] = alo[i]; a[mi][i + 8] = ahi[i]; }
    }
#pragma unroll
    for (int ni = 0; ni < 2; ni++)
      // B 32x16: lane holds one column (N), K range = k0 + hl*16 .. +16
      b[ni] = *(const v16h*)(Brow[ni] + k0);
#pragma unroll
    for (int mi = 0; mi < 2; mi++)
#pragma unroll
      for (int ni = 0; ni < 2; ni++)
        acc[mi][ni] = __builtin_amdgcn_wmma_f32_16x16x32_f16(
            false, a[mi], false, b[ni], (short)0, acc[mi][ni], false, false);
  }
#pragma unroll
  for (int ni = 0; ni < 2; ni++){
    int col = nbase + ni * 16 + l16;
    if (col >= N) continue;
    float bv = bias ? bias[col] : 0.f;
#pragma unroll
    for (int mi = 0; mi < 2; mi++){
#pragma unroll
      for (int v = 0; v < 8; v++){
        int row = mbase + mi * 16 + v + hl * 8;  // C/D layout: VGPR v -> M=v+8*half
        float val = acc[mi][ni][v] + bv;
        if (Cin) val += Cin[(size_t)row * N + col];
        if (act_gelu) val = 0.5f * val * (1.f + erff(val * 0.70710678118f));
        if (outF) outF[(size_t)row * N + col] = val;
        else      outH[(size_t)row * N + col] = (_Float16)val;
      }
    }
  }
}

// ---------------------------------------------------------------------------
// Depthwise causal conv(4) + SiLU + mask; dt = softplus(dtr+bias)*mask;
// a = exp(-exp(A_log)*dt); xdt = xconv*dt
// ---------------------------------------------------------------------------
__global__ void k_conv(const float* __restrict__ proj, const float* __restrict__ cw,
                       const float* __restrict__ A_log, const float* __restrict__ dt_bias,
                       const unsigned char* __restrict__ mask,
                       float* __restrict__ xh, float* __restrict__ xdt,
                       float* __restrict__ abuf){
  int row = blockIdx.x;            // b*AA + t
  int t = row & (AA - 1);
  int c = threadIdx.x;             // 0..511
  float mv = mask[row] ? 1.f : 0.f;
  float s = 0.f;
#pragma unroll
  for (int k = 0; k < DCONV; k++){
    int tt = t - (DCONV - 1) + k;
    if (tt >= 0)
      s += cw[k * DI + c] * proj[(size_t)(row - (DCONV - 1) + k) * DPROJ + DI + c];
  }
  float xv = siluf(s) * mv;
  xh[(size_t)row * DI + c] = xv;
  int h = c >> 6;
  float dr = proj[(size_t)row * DPROJ + 2 * DI + 2 * RR * NSTATE + h] + dt_bias[h];
  float dt = (dr > 20.f ? dr : log1pf(expf(dr))) * mv;
  xdt[(size_t)row * DI + c] = xv * dt;
  if ((c & 63) == 0) abuf[(size_t)row * NH + h] = expf(-expf(A_log[h]) * dt);
}

// ---------------------------------------------------------------------------
// Selective MIMO scan, one block per (head, batch, direction).
// Thread t: p = t&63, handles n in [ (t>>6)*8, +8 ) for r=0..1 -> 16 states in VGPRs.
// ---------------------------------------------------------------------------
__global__ void k_scan(const float* __restrict__ proj, const float* __restrict__ xdt,
                       const float* __restrict__ abuf,
                       float* __restrict__ yf, float* __restrict__ yr){
  __shared__ float sB[RR * NSTATE];
  __shared__ float sC[RR * NSTATE];
  __shared__ float sX[PP];
  __shared__ float sA;
  __shared__ float sY[256];
  int h = blockIdx.x, b = blockIdx.y, dir = blockIdx.z;
  float* yout = dir ? yr : yf;
  int t = threadIdx.x;
  int p = t & 63, nb = t >> 6;
  float hs[RR][8];
#pragma unroll
  for (int r = 0; r < RR; r++)
#pragma unroll
    for (int j = 0; j < 8; j++) hs[r][j] = 0.f;

  for (int s = 0; s < AA; s++){
    int tt = dir ? (AA - 1 - s) : s;
    size_t row = (size_t)b * AA + tt;
    if (t < 64)        sB[t]       = proj[row * DPROJ + 2 * DI + t];
    else if (t < 128)  sC[t - 64]  = proj[row * DPROJ + 2 * DI + RR * NSTATE + (t - 64)];
    else if (t < 192)  sX[t - 128] = xdt[row * DI + h * PP + (t - 128)];
    else if (t == 192) sA          = abuf[row * NH + h];
    __syncthreads();
    float av = sA, xv = sX[p], yp = 0.f;
#pragma unroll
    for (int r = 0; r < RR; r++){
#pragma unroll
      for (int j = 0; j < 8; j++){
        int n = nb * 8 + j;
        float hh = av * hs[r][j] + sB[r * NSTATE + n] * xv;
        hs[r][j] = hh;
        yp += hh * sC[r * NSTATE + n];
      }
    }
    sY[t] = yp;
    __syncthreads();
    if (t < 64)
      yout[row * DI + h * PP + t] = sY[t] + sY[t + 64] + sY[t + 128] + sY[t + 192];
  }
}

// ---------------------------------------------------------------------------
// y_total = yf + yr + Dp*xh; g = y_total*silu(z); RMSNorm(g)*rms_w -> f16
// ---------------------------------------------------------------------------
__global__ void k_gate(const float* __restrict__ proj, const float* __restrict__ xh,
                       const float* __restrict__ yf, const float* __restrict__ yr,
                       const float* __restrict__ Dp, const float* __restrict__ rw,
                       _Float16* __restrict__ out){
  __shared__ float red[256];
  int row = blockIdx.x, t = threadIdx.x;
  float g[2]; float ss = 0.f;
#pragma unroll
  for (int i = 0; i < 2; i++){
    int c = t + i * 256;
    size_t idx = (size_t)row * DI + c;
    float yt = yf[idx] + yr[idx] + Dp[c >> 6] * xh[idx];
    float gv = yt * siluf(proj[(size_t)row * DPROJ + c]);   // z slice
    g[i] = gv; ss += gv * gv;
  }
  red[t] = ss; __syncthreads();
  for (int s = 128; s > 0; s >>= 1){ if (t < s) red[t] += red[t + s]; __syncthreads(); }
  float scale = rsqrtf(red[0] / (float)DI + 1e-6f);
#pragma unroll
  for (int i = 0; i < 2; i++){
    int c = t + i * 256;
    out[(size_t)row * DI + c] = (_Float16)(g[i] * scale * rw[c]);
  }
}

// ---------------------------------------------------------------------------
// Head LayerNorm -> f16
// ---------------------------------------------------------------------------
__global__ void k_ln16(const float* __restrict__ x, const float* __restrict__ g,
                       const float* __restrict__ bta, _Float16* __restrict__ out){
  __shared__ float red[256];
  int row = blockIdx.x, t = threadIdx.x;
  float v = x[(size_t)row * DA + t];
  red[t] = v; __syncthreads();
  for (int s = 128; s > 0; s >>= 1){ if (t < s) red[t] += red[t + s]; __syncthreads(); }
  float mu = red[0] / (float)DA;
  __syncthreads();
  float d = v - mu;
  red[t] = d * d; __syncthreads();
  for (int s = 128; s > 0; s >>= 1){ if (t < s) red[t] += red[t + s]; __syncthreads(); }
  float inv = rsqrtf(red[0] / (float)DA + 1e-5f);
  out[(size_t)row * DA + t] = (_Float16)(d * inv * g[t] + bta[t]);
}

// ---------------------------------------------------------------------------
// Final tiny GEMM: [8192,128](f16) @ [128,3] + b2, masked
// ---------------------------------------------------------------------------
__global__ void k_head2(const _Float16* __restrict__ h2, const float* __restrict__ w2,
                        const float* __restrict__ b2, const unsigned char* __restrict__ mask,
                        float* __restrict__ out){
  __shared__ float red[3][128];
  int row = blockIdx.x, t = threadIdx.x;
  float hv = (float)h2[(size_t)row * 128 + t];
#pragma unroll
  for (int o = 0; o < 3; o++) red[o][t] = hv * w2[t * 3 + o];
  __syncthreads();
  for (int s = 64; s > 0; s >>= 1){
    if (t < s){ red[0][t] += red[0][t+s]; red[1][t] += red[1][t+s]; red[2][t] += red[2][t+s]; }
    __syncthreads();
  }
  if (t < 3) out[(size_t)row * 3 + t] = (red[t][0] + b2[t]) * (mask[row] ? 1.f : 0.f);
}

// ---------------------------------------------------------------------------
extern "C" void kernel_launch(void* const* d_in, const int* in_sizes, int n_in,
                              void* d_out, int out_size, void* d_ws, size_t ws_size,
                              hipStream_t stream) {
  (void)in_sizes; (void)n_in; (void)out_size; (void)ws_size;
  const float*         tok_in  = (const float*)d_in[0];
  const unsigned char* mask    = (const unsigned char*)d_in[1];
  const float*         w_in    = (const float*)d_in[2];
  const float*         conv_w  = (const float*)d_in[3];
  const float*         A_log   = (const float*)d_in[4];
  const float*         dt_bias = (const float*)d_in[5];
  const float*         Dp      = (const float*)d_in[6];
  const float*         rms_w   = (const float*)d_in[7];
  const float*         w_out   = (const float*)d_in[8];
  const float*         pre_w   = (const float*)d_in[9];
  const float*         ln_g    = (const float*)d_in[10];
  const float*         ln_b    = (const float*)d_in[11];
  const float*         w1      = (const float*)d_in[12];
  const float*         b1      = (const float*)d_in[13];
  const float*         w2      = (const float*)d_in[14];
  const float*         b2      = (const float*)d_in[15];

  char* base = (char*)d_ws;
  size_t off = 0;
  auto alloc = [&](size_t bytes)->void* {
    void* p = base + off; off += (bytes + 255) & ~(size_t)255; return p;
  };
  float*    tok0  = (float*)alloc((size_t)MM * DA * 4);
  float*    tok1  = (float*)alloc((size_t)MM * DA * 4);
  _Float16* xn16  = (_Float16*)alloc((size_t)MM * DA * 2);
  _Float16* wInT  = (_Float16*)alloc((size_t)4 * NPAD_IN * DA * 2);
  _Float16* wOutT = (_Float16*)alloc((size_t)4 * DA * DI * 2);
  _Float16* w1T   = (_Float16*)alloc((size_t)128 * DA * 2);
  float*    proj  = (float*)alloc((size_t)MM * DPROJ * 4);
  float*    xh    = (float*)alloc((size_t)MM * DI * 4);
  float*    xdt   = (float*)alloc((size_t)MM * DI * 4);
  float*    abuf  = (float*)alloc((size_t)MM * NH * 4);
  float*    yf    = (float*)alloc((size_t)MM * DI * 4);
  float*    yr    = (float*)alloc((size_t)MM * DI * 4);
  _Float16* g16   = (_Float16*)alloc((size_t)MM * DI * 2);
  _Float16* hln16 = (_Float16*)alloc((size_t)MM * DA * 2);
  _Float16* h2_16 = (_Float16*)alloc((size_t)MM * 128 * 2);

  // Weight convert + transpose to f16 (per call; deterministic)
  for (int l = 0; l < 4; l++){
    k_w2h_t<<<(NPAD_IN * DA + 255) / 256, 256, 0, stream>>>(
        w_in + (size_t)l * DA * DPROJ, wInT + (size_t)l * NPAD_IN * DA, DA, DPROJ, NPAD_IN);
    k_w2h_t<<<(DA * DI + 255) / 256, 256, 0, stream>>>(
        w_out + (size_t)l * DI * DA, wOutT + (size_t)l * DA * DI, DI, DA, DA);
  }
  k_w2h_t<<<(128 * DA + 255) / 256, 256, 0, stream>>>(w1, w1T, DA, 128, 128);

  hipMemcpyAsync(tok0, tok_in, (size_t)MM * DA * 4, hipMemcpyDeviceToDevice, stream);

  float* tokA = tok0; float* tokB = tok1;
  for (int l = 0; l < 4; l++){
    k_rms16<<<MM, 256, 0, stream>>>(tokA, pre_w + l * DA, xn16);
    // proj = x_norm @ w_in    [8192 x 1160 x 256]
    k_wmma_gemm<<<dim3(NPAD_IN / 32, MM / 32), 32, 0, stream>>>(
        xn16, wInT + (size_t)l * NPAD_IN * DA, MM, DPROJ, DA,
        nullptr, nullptr, proj, nullptr, 0);
    k_conv<<<MM, DI, 0, stream>>>(proj, conv_w + (size_t)l * DCONV * DI,
                                  A_log + l * NH, dt_bias + l * NH, mask, xh, xdt, abuf);
    k_scan<<<dim3(NH, BB, 2), 256, 0, stream>>>(proj, xdt, abuf, yf, yr);
    k_gate<<<MM, 256, 0, stream>>>(proj, xh, yf, yr, Dp + l * NH, rms_w + l * DI, g16);
    // tokB = tokA + g @ w_out  [8192 x 256 x 512]
    k_wmma_gemm<<<dim3(DA / 32, MM / 32), 32, 0, stream>>>(
        g16, wOutT + (size_t)l * DA * DI, MM, DA, DI,
        tokA, nullptr, tokB, nullptr, 0);
    float* tmp = tokA; tokA = tokB; tokB = tmp;
  }

  // Head: LN -> GEMM(+b1, GELU) -> tiny GEMM(+b2, mask)
  k_ln16<<<MM, 256, 0, stream>>>(tokA, ln_g, ln_b, hln16);
  k_wmma_gemm<<<dim3(128 / 32, MM / 32), 32, 0, stream>>>(
      hln16, w1T, MM, 128, DA, nullptr, b1, nullptr, h2_16, 1);
  k_head2<<<MM, 128, 0, stream>>>(h2_16, w2, b2, mask, (float*)d_out);
}
